// ShiftedWindowAttention_43387759624740
// MI455X (gfx1250) — matrix operand order repaired and checked by reading, not compiled, not run on hardware
//
#include <hip/hip_runtime.h>

// ---------------------------------------------------------------------------
// CDNA5 (gfx1250, wave32) fused Swin shifted-window attention.
// All GEMMs via v_wmma_f32_16x16x32_f16 (f16 inputs, f32 accumulate).
// One workgroup (8 waves) per 64-token window; all 8 heads' scores kept in
// LDS simultaneously so stage-2 needs only 3 barriers total.
// ---------------------------------------------------------------------------

typedef __attribute__((ext_vector_type(16))) _Float16 v16h;
typedef __attribute__((ext_vector_type(8)))  float    v8f;
typedef _Float16 h16;

#define NB      32
#define HIMG    64
#define WIMG    64
#define CCH     256
#define WSZ     8
#define SSH     4
#define NHEADS  8
#define HDIM    32
#define WS2     64
#define NWIN    64
#define SCALE_F 0.17677669529663687f   // 32^-0.5

// LDS row strides (elements). All multiples of 8 -> 16B-aligned b128 frags,
// chosen so row-base bank index is distinct across the 16 lanes of a frag.
#define XW_S   264   // x window / attn-out buffer, f16 [64][264]
#define QK_S   40    // q,k per head, f16 [64][40]
#define VT_S   72    // v transposed per head, f16 [32][72]
#define AT_S   68    // attn scores f32 [64][68] per head
#define AT_SH  136   // same rows viewed as f16 [64][136]

#define WMMA_F16(a, b, c) \
  __builtin_amdgcn_wmma_f32_16x16x32_f16(false, (a), false, (b), (short)0, (c), false, false)

// A-fragment (16x32 f16): lane<16 holds row m, K = k0+{0..7, 16..23};
// lane>=16 holds row m, K = k0+{8..15, 24..31}. Two 16B loads.
__device__ __forceinline__ v16h load_a_frag(const h16* base, int row0, int stride,
                                            int k0, int lane) {
  const h16* p = base + (row0 + (lane & 15)) * stride + k0 + ((lane & 16) ? 8 : 0);
  union { v16h h; uint4 u[2]; } t;
  t.u[0] = *(const uint4*)p;
  t.u[1] = *(const uint4*)(p + 16);
  return t.h;
}

// B-fragment (32x16 f16) loaded from B^T stored row-major [N][K]:
// lane<16 -> col n, K = k0+{0..7, 16..23} ; lane>=16 -> col n, K = k0+{8..15, 24..31}.
__device__ __forceinline__ v16h load_bt_frag(const h16* baseT, int n0, int strideT,
                                             int k0, int lane) {
  const h16* p = baseT + (n0 + (lane & 15)) * strideT + k0 + ((lane & 16) ? 8 : 0);
  union { v16h h; uint4 u[2]; } t;
  t.u[0] = *(const uint4*)p;
  t.u[1] = *(const uint4*)(p + 16);
  return t.h;
}

// ---------------------------------------------------------------------------
// Prep: f32 -> f16 transposed weights + relative-position bias table.
// ---------------------------------------------------------------------------
__global__ void swin_prep_kernel(const float* __restrict__ w_qkv,
                                 const float* __restrict__ w_out,
                                 const float* __restrict__ pos_enc,
                                 h16* __restrict__ wqkvT,
                                 h16* __restrict__ woutT,
                                 float* __restrict__ biasTab) {
  int i = blockIdx.x * blockDim.x + threadIdx.x;
  const int N1 = 768 * 256;
  const int N2 = 256 * 256;
  const int N3 = NHEADS * WS2 * WS2;
  if (i < N1) {                       // wqkvT[n][k] = w_qkv[k][n]
    int n = i >> 8, k = i & 255;
    wqkvT[i] = (h16)w_qkv[k * 768 + n];
  } else if (i < N1 + N2) {           // woutT[n][k] = w_out[k][n]
    int j = i - N1;
    int n = j >> 8, k = j & 255;
    woutT[j] = (h16)w_out[k * 256 + n];
  } else if (i < N1 + N2 + N3) {      // biasTab[h][q][kk]
    int j = i - (N1 + N2);
    int h = j >> 12;
    int q = (j >> 6) & 63, kk = j & 63;
    int dy = (q >> 3) - (kk >> 3) + (WSZ - 1);
    int dx = (q & 7) - (kk & 7) + (WSZ - 1);
    biasTab[j] = pos_enc[h * (2 * WSZ - 1) * (2 * WSZ - 1) + dy * (2 * WSZ - 1) + dx];
  }
}

// ---------------------------------------------------------------------------
// Fused attention kernel.
// ---------------------------------------------------------------------------
__global__ __launch_bounds__(256, 1)
void swin_attn_kernel(const float* __restrict__ x,
                      const float* __restrict__ b_qkv,
                      const float* __restrict__ b_out,
                      const h16*  __restrict__ wqkvT,
                      const h16*  __restrict__ woutT,
                      const float* __restrict__ biasTab,
                      float* __restrict__ out) {
  __shared__ __align__(16) h16   s_xw[WS2 * XW_S];            // x tile, later attn output (A side)
  __shared__ __align__(16) h16   s_q[NHEADS * WS2 * QK_S];    // [h][row][hd]
  __shared__ __align__(16) h16   s_k[NHEADS * WS2 * QK_S];    // [h][key][hd]  (== B^T for QK^T)
  __shared__ __align__(16) h16   s_vT[NHEADS * HDIM * VT_S];  // [h][hd][key]  (== B^T for PV)
  __shared__ __align__(16) float s_attn[NHEADS * WS2 * AT_S]; // f32 scores / f16 probs, all heads

  const int tid  = threadIdx.x;
  const int lane = tid & 31;
  const int wave = tid >> 5;
  const int blk  = blockIdx.x;
  const int b    = blk >> 6;
  const int wy   = (blk >> 3) & 7;
  const int wx   = blk & 7;
  const int hi8  = (lane & 16) ? 8 : 0;
  const int l15  = lane & 15;

  // ---- stage 0: gather rolled window into LDS as f16 ----
  {
    const size_t xbase = (size_t)b * (HIMG * WIMG) * CCH;
    for (int i = tid; i < WS2 * CCH; i += 256) {
      int row = i >> 8;
      int col = i & 255;
      int oy = (wy * WSZ + (row >> 3) + SSH) & (HIMG - 1);
      int ox = (wx * WSZ + (row & 7) + SSH) & (WIMG - 1);
      s_xw[row * XW_S + col] = (h16)x[xbase + (size_t)(oy * WIMG + ox) * CCH + col];
    }
  }
  __syncthreads();

  // ---- stage 1: qkv = xw @ w_qkv + b_qkv  (64x768, K=256) ----
  for (int nt = wave; nt < 48; nt += 8) {
    const int coln = nt * 16 + l15;
    const float bv = b_qkv[coln];
    v8f acc[4];
#pragma unroll
    for (int mt = 0; mt < 4; ++mt)
#pragma unroll
      for (int r = 0; r < 8; ++r) acc[mt][r] = bv;
#pragma unroll
    for (int k0 = 0; k0 < 256; k0 += 32) {
      v16h bf = load_bt_frag(wqkvT, nt * 16, 256, k0, lane);
#pragma unroll
      for (int mt = 0; mt < 4; ++mt) {
        v16h af = load_a_frag(s_xw, mt * 16, XW_S, k0, lane);
        acc[mt] = WMMA_F16(af, bf, acc[mt]);
      }
    }
    // scatter D tiles: q/k row-major per head, v transposed per head
#pragma unroll
    for (int mt = 0; mt < 4; ++mt) {
      if (coln < 256) {
#pragma unroll
        for (int r = 0; r < 8; ++r) {
          int row = mt * 16 + r + hi8;
          s_q[((coln >> 5) * WS2 + row) * QK_S + (coln & 31)] = (h16)acc[mt][r];
        }
      } else if (coln < 512) {
        int c2 = coln - 256;
#pragma unroll
        for (int r = 0; r < 8; ++r) {
          int row = mt * 16 + r + hi8;
          s_k[((c2 >> 5) * WS2 + row) * QK_S + (c2 & 31)] = (h16)acc[mt][r];
        }
      } else {
        int c2 = coln - 512;
        // transposed store: rows are contiguous -> pack pairs into u32 stores
        int base = ((c2 >> 5) * HDIM + (c2 & 31)) * VT_S + mt * 16 + hi8;
#pragma unroll
        for (int r = 0; r < 8; r += 2) {
          union { unsigned u; h16 hh[2]; } t;
          t.hh[0] = (h16)acc[mt][r];
          t.hh[1] = (h16)acc[mt][r + 1];
          *(unsigned*)(&s_vT[base + r]) = t.u;
        }
      }
    }
  }
  __syncthreads();

  // ---- stage 2a: S_h = scale * (q_h k_h^T) + bias_h, masked — ALL heads ----
  {
    const int mt = wave & 3;
    for (int h = 0; h < NHEADS; ++h) {
      const h16* qh = s_q + h * WS2 * QK_S;
      const h16* kh = s_k + h * WS2 * QK_S;
      float* ah = s_attn + h * WS2 * AT_S;
      v16h af = load_a_frag(qh, mt * 16, QK_S, 0, lane);
#pragma unroll
      for (int j = 0; j < 2; ++j) {
        int nt = (wave >> 2) + 2 * j;
        v16h bf = load_bt_frag(kh, nt * 16, QK_S, 0, lane);
        v8f acc = {};
        acc = WMMA_F16(af, bf, acc);
        int colk = nt * 16 + l15;
        int ky = colk >> 3, kx = colk & 7;
#pragma unroll
        for (int r = 0; r < 8; ++r) {
          int row = mt * 16 + r + hi8;
          int qy = row >> 3, qx = row & 7;
          bool maskd = ((wy == 7) && ((qy >= SSH) != (ky >= SSH))) ||
                       ((wx == 7) && ((qx >= SSH) != (kx >= SSH)));
          float v = acc[r] * SCALE_F + biasTab[(h * WS2 + row) * WS2 + colk];
          ah[row * AT_S + colk] = maskd ? -1.0e30f : v;
        }
      }
    }
  }
  __syncthreads();

  // ---- stage 2b: row softmax over all 512 rows; 2 rows per thread.
  //      Emit f16 probabilities in-place (f16 row stride 136), packed 16B.
  for (int rr = tid; rr < NHEADS * WS2; rr += 256) {
    const int h = rr >> 6;
    const int r = rr & 63;
    const float* rp = s_attn + h * (WS2 * AT_S) + r * AT_S;
    float vals[WS2];
#pragma unroll
    for (int j = 0; j < WS2; ++j) vals[j] = rp[j];
    float m = vals[0];
#pragma unroll
    for (int j = 1; j < WS2; ++j) m = fmaxf(m, vals[j]);
    float s = 0.f;
#pragma unroll
    for (int j = 0; j < WS2; ++j) { vals[j] = __expf(vals[j] - m); s += vals[j]; }
    float inv = 1.f / s;
    h16* wp = (h16*)(s_attn + h * (WS2 * AT_S)) + r * AT_SH;
#pragma unroll
    for (int j = 0; j < WS2; j += 8) {
      union { uint4 u; h16 hh[8]; } t;
#pragma unroll
      for (int q2 = 0; q2 < 8; ++q2) t.hh[q2] = (h16)(vals[j + q2] * inv);
      *(uint4*)(wp + j) = t.u;
    }
  }
  __syncthreads();

  // ---- stage 2c: O_h = P_h @ V_h (64x32, K=64) — ALL heads ----
  {
    const int mt = wave & 3;
    const int nt = wave >> 2;   // 0..1
    for (int h = 0; h < NHEADS; ++h) {
      const h16* vTh = s_vT + h * HDIM * VT_S;
      const h16* ph  = (const h16*)(s_attn + h * (WS2 * AT_S));
      v8f acc = {};
#pragma unroll
      for (int k0 = 0; k0 < WS2; k0 += 32) {
        v16h af = load_a_frag(ph, mt * 16, AT_SH, k0, lane);
        v16h bf = load_bt_frag(vTh, nt * 16, VT_S, k0, lane);
        acc = WMMA_F16(af, bf, acc);
      }
      int cc = h * HDIM + nt * 16 + l15;
#pragma unroll
      for (int r = 0; r < 8; ++r) {
        int row = mt * 16 + r + hi8;
        s_xw[row * XW_S + cc] = (h16)acc[r];
      }
    }
  }
  __syncthreads();

  // ---- stage 3: out = av @ w_out + b_out, scatter with reverse roll ----
  for (int nt = wave; nt < 16; nt += 8) {
    const int coln = nt * 16 + l15;
    const float bv = b_out[coln];
    v8f acc[4];
#pragma unroll
    for (int mt = 0; mt < 4; ++mt)
#pragma unroll
      for (int r = 0; r < 8; ++r) acc[mt][r] = bv;
#pragma unroll
    for (int k0 = 0; k0 < 256; k0 += 32) {
      v16h bf = load_bt_frag(woutT, nt * 16, 256, k0, lane);
#pragma unroll
      for (int mt = 0; mt < 4; ++mt) {
        v16h af = load_a_frag(s_xw, mt * 16, XW_S, k0, lane);
        acc[mt] = WMMA_F16(af, bf, acc[mt]);
      }
    }
#pragma unroll
    for (int mt = 0; mt < 4; ++mt) {
#pragma unroll
      for (int r = 0; r < 8; ++r) {
        int row = mt * 16 + r + hi8;
        int oy = (wy * WSZ + (row >> 3) + SSH) & (HIMG - 1);
        int ox = (wx * WSZ + (row & 7) + SSH) & (WIMG - 1);
        out[((size_t)b * (HIMG * WIMG) + oy * WIMG + ox) * CCH + coln] = acc[mt][r];
      }
    }
  }
}

// ---------------------------------------------------------------------------
extern "C" void kernel_launch(void* const* d_in, const int* in_sizes, int n_in,
                              void* d_out, int out_size, void* d_ws, size_t ws_size,
                              hipStream_t stream) {
  const float* x       = (const float*)d_in[0];
  const float* w_qkv   = (const float*)d_in[1];
  const float* b_qkv   = (const float*)d_in[2];
  const float* w_out   = (const float*)d_in[3];
  const float* b_out   = (const float*)d_in[4];
  const float* pos_enc = (const float*)d_in[5];
  float* out = (float*)d_out;

  // workspace layout: wqkvT f16 [768*256] | woutT f16 [256*256] | biasTab f32 [8*64*64]
  h16*   wqkvT   = (h16*)d_ws;
  h16*   woutT   = wqkvT + 768 * 256;
  float* biasTab = (float*)((char*)d_ws + (size_t)(768 * 256 + 256 * 256) * sizeof(h16));

  const int prep_elems = 768 * 256 + 256 * 256 + NHEADS * WS2 * WS2;
  swin_prep_kernel<<<(prep_elems + 255) / 256, 256, 0, stream>>>(
      w_qkv, w_out, pos_enc, wqkvT, woutT, biasTab);

  swin_attn_kernel<<<NB * NWIN, 256, 0, stream>>>(
      x, b_qkv, b_out, wqkvT, woutT, biasTab, out);
}
